// DiagSSMBlock_76347338654364
// MI455X (gfx1250) — matrix operand order repaired and checked by reading, not compiled, not run on hardware
//
#include <hip/hip_runtime.h>
#include <hip/hip_bf16.h>

// ---------------------------------------------------------------------------
// DiagSSM: out[t,h] = sum_{k<=t} a[h]^(t-k) * (x @ B)[k,h]
//   Stage 1: S = x @ B  (4096x2048x2048) via bf16 WMMA, f32 accumulate
//   Stage 2: per-channel first-order scan h_t = a*h_{t-1} + s_t (chunked)
// ---------------------------------------------------------------------------

typedef __bf16 bf16;
typedef __attribute__((ext_vector_type(16))) __bf16 v16bf;
typedef __attribute__((ext_vector_type(8)))  __bf16 v8bf;
typedef __attribute__((ext_vector_type(4)))  __bf16 v4bf;
typedef __attribute__((ext_vector_type(8)))  float  v8f;

#define TDIM 4096
#define HDIM 2048
#define BM 128          // block tile M (rows of x / S)
#define BN 128          // block tile N (cols of B / S)
#define KCHUNK 64       // K staged in LDS per iteration
#define LDS_PAD 8       // bf16 elements of row padding (16B -> conflict-free b128)
#define LDS_STRIDE (KCHUNK + LDS_PAD)

#define SCAN_C 64       // number of time chunks
#define SCAN_L 64       // chunk length (SCAN_C * SCAN_L == TDIM)

// Combine two 16B-aligned 8-element bf16 LDS reads into one 16-element fragment.
__device__ __forceinline__ v16bf frag_combine(const bf16* lo, const bf16* hi) {
    v8bf l = *(const v8bf*)lo;
    v8bf h = *(const v8bf*)hi;
    v16bf r;
#pragma unroll
    for (int i = 0; i < 8; ++i) { r[i] = l[i]; r[i + 8] = h[i]; }
    return r;
}

// ---------------------------------------------------------------------------
// GEMM: S[t,k] = sum_h x[t,h] * B[h,k]   (row-major, f32 in/out, bf16 WMMA core)
// grid = (HDIM/BN, TDIM/BM), block = 256 threads (8 waves)
// ---------------------------------------------------------------------------
__global__ __launch_bounds__(256) void diagssm_gemm_bf16(
    const float* __restrict__ x, const float* __restrict__ Bmat,
    float* __restrict__ S)
{
    __shared__ bf16 As[BM][LDS_STRIDE];   // [m][k] row-major in k
    __shared__ bf16 Bs[BN][LDS_STRIDE];   // [n][k] transposed: k contiguous

    const int tid      = threadIdx.x;
    const int wave     = tid >> 5;
    const int lane     = tid & 31;
    const int laneRow  = lane & 15;
    const int laneHalf = lane >> 4;
    const int waveM    = (wave & 3) * 32;   // 4 waves down M
    const int waveN    = (wave >> 2) * 64;  // 2 waves across N
    const int blockM   = blockIdx.y * BM;
    const int blockN   = blockIdx.x * BN;

    v8f acc[2][4];
#pragma unroll
    for (int mt = 0; mt < 2; ++mt)
#pragma unroll
        for (int nt = 0; nt < 4; ++nt)
            acc[mt][nt] = (v8f){0.f, 0.f, 0.f, 0.f, 0.f, 0.f, 0.f, 0.f};

    // staging thread mapping
    const int xm0 = tid >> 4;            // 0..15, step 16 -> rows of x tile
    const int xk0 = (tid & 15) * 4;      // float4 column within K chunk
    const int bk0 = tid >> 5;            // 0..7, step 8 -> rows (k) of B tile
    const int bn0 = (tid & 31) * 4;      // float4 column within N tile

    for (int kb = 0; kb < HDIM; kb += KCHUNK) {
        // ---- stage x chunk (BM x KCHUNK f32 -> bf16) ----
#pragma unroll
        for (int i = 0; i < 8; ++i) {
            const int m = xm0 + i * 16;
            const float4 v = *(const float4*)(x + (size_t)(blockM + m) * HDIM + kb + xk0);
            v4bf w;
            w[0] = (bf16)v.x; w[1] = (bf16)v.y; w[2] = (bf16)v.z; w[3] = (bf16)v.w;
            *(v4bf*)&As[m][xk0] = w;
        }
        // ---- stage B chunk (KCHUNK x BN f32 -> bf16, transposed into LDS) ----
#pragma unroll
        for (int i = 0; i < 8; ++i) {
            const int kk = bk0 + i * 8;
            const float4 v = *(const float4*)(Bmat + (size_t)(kb + kk) * HDIM + blockN + bn0);
            Bs[bn0 + 0][kk] = (bf16)v.x;
            Bs[bn0 + 1][kk] = (bf16)v.y;
            Bs[bn0 + 2][kk] = (bf16)v.z;
            Bs[bn0 + 3][kk] = (bf16)v.w;
        }
        __syncthreads();

        // prefetch next K chunk while computing (global_prefetch_b8)
        if (kb + KCHUNK < HDIM) {
            __builtin_prefetch(x + (size_t)(blockM + xm0) * HDIM + kb + KCHUNK, 0, 0);
            __builtin_prefetch(Bmat + (size_t)(kb + KCHUNK + bk0) * HDIM + blockN, 0, 0);
        }

        // ---- compute: two K=32 WMMA steps over the staged chunk ----
#pragma unroll
        for (int ks = 0; ks < KCHUNK; ks += 32) {
            v16bf afrag[2];
#pragma unroll
            for (int mt = 0; mt < 2; ++mt) {
                const int row = waveM + mt * 16 + laneRow;
                const int ka  = ks + laneHalf * 8;     // lanes 0-15: K 0-7/16-23; 16-31: 8-15/24-31
                afrag[mt] = frag_combine(&As[row][ka], &As[row][ka + 16]);
            }
            v16bf bfrag[4];
#pragma unroll
            for (int nt = 0; nt < 4; ++nt) {
                const int col = waveN + nt * 16 + laneRow;
                const int kbf = ks + laneHalf * 8;     // lanes 0-15: K 0-7/16-23; 16-31: 8-15/24-31
                bfrag[nt] = frag_combine(&Bs[col][kbf], &Bs[col][kbf + 16]);
            }
#pragma unroll
            for (int mt = 0; mt < 2; ++mt)
#pragma unroll
                for (int nt = 0; nt < 4; ++nt)
                    acc[mt][nt] = __builtin_amdgcn_wmma_f32_16x16x32_bf16(
                        false, afrag[mt], false, bfrag[nt],
                        (short)0, acc[mt][nt], false, false);
        }
        __syncthreads();
    }

    // ---- epilogue: C/D layout -> row = r + laneHalf*8, col = laneRow ----
#pragma unroll
    for (int mt = 0; mt < 2; ++mt) {
#pragma unroll
        for (int nt = 0; nt < 4; ++nt) {
            const int col = blockN + waveN + nt * 16 + laneRow;
#pragma unroll
            for (int r = 0; r < 8; ++r) {
                const int row = blockM + waveM + mt * 16 + r + laneHalf * 8;
                S[(size_t)row * HDIM + col] = acc[mt][nt][r];
            }
        }
    }
}

// ---------------------------------------------------------------------------
// Scan phase 1: per (chunk c, channel h) local scan end-state f[c,h]
// grid = (HDIM/256, SCAN_C), block = 256
// ---------------------------------------------------------------------------
__global__ __launch_bounds__(256) void diagssm_scan1(
    const float* __restrict__ S, const float* __restrict__ a,
    float* __restrict__ f)
{
    const int h = blockIdx.x * blockDim.x + threadIdx.x;
    const int c = blockIdx.y;
    const float av = a[h];
    const float* p = S + (size_t)c * SCAN_L * HDIM + h;
    float acc = 0.f;
#pragma unroll 4
    for (int t = 0; t < SCAN_L; ++t) { acc = fmaf(av, acc, *p); p += HDIM; }
    f[(size_t)c * HDIM + h] = acc;
}

// ---------------------------------------------------------------------------
// Scan phase 2: scan the chunk states: g[c] = a^L * g[c-1] + f[c] (in-place)
// grid = (HDIM/256), block = 256
// ---------------------------------------------------------------------------
__global__ __launch_bounds__(256) void diagssm_scan2(
    const float* __restrict__ a, float* __restrict__ f)
{
    const int h = blockIdx.x * blockDim.x + threadIdx.x;
    const float av = a[h];
    float aL = av;
#pragma unroll
    for (int i = 0; i < 6; ++i) aL = aL * aL;   // a^64
    float g = 0.f;
#pragma unroll 4
    for (int c = 0; c < SCAN_C; ++c) {
        g = fmaf(aL, g, f[(size_t)c * HDIM + h]);
        f[(size_t)c * HDIM + h] = g;
    }
}

// ---------------------------------------------------------------------------
// Scan phase 3: replay each chunk with carry-in = f[c-1,h], write output
// grid = (HDIM/256, SCAN_C), block = 256
// ---------------------------------------------------------------------------
__global__ __launch_bounds__(256) void diagssm_scan3(
    const float* __restrict__ S, const float* __restrict__ a,
    const float* __restrict__ f, float* __restrict__ out)
{
    const int h = blockIdx.x * blockDim.x + threadIdx.x;
    const int c = blockIdx.y;
    const float av = a[h];
    float acc = (c == 0) ? 0.f : f[(size_t)(c - 1) * HDIM + h];
    const float* p = S + (size_t)c * SCAN_L * HDIM + h;
    float* q = out + (size_t)c * SCAN_L * HDIM + h;
#pragma unroll 4
    for (int t = 0; t < SCAN_L; ++t) {
        acc = fmaf(av, acc, *p);
        *q = acc;
        p += HDIM; q += HDIM;
    }
}

extern "C" void kernel_launch(void* const* d_in, const int* in_sizes, int n_in,
                              void* d_out, int out_size, void* d_ws, size_t ws_size,
                              hipStream_t stream) {
    const float* x    = (const float*)d_in[0];   // (T, H) f32
    const float* a    = (const float*)d_in[1];   // (H,)  f32
    const float* Bmat = (const float*)d_in[2];   // (H, H) f32
    float*       out  = (float*)d_out;           // (1, T, H) f32

    float* S = (float*)d_ws;                     // T*H f32 = 32 MB
    float* f = S + (size_t)TDIM * HDIM;          // SCAN_C*H f32 = 512 KB

    dim3 gGemm(HDIM / BN, TDIM / BM);            // (16, 32)
    diagssm_gemm_bf16<<<gGemm, 256, 0, stream>>>(x, Bmat, S);

    dim3 gScan(HDIM / 256, SCAN_C);              // (8, 64)
    diagssm_scan1<<<gScan, 256, 0, stream>>>(S, a, f);
    diagssm_scan2<<<dim3(HDIM / 256), 256, 0, stream>>>(a, f);
    diagssm_scan3<<<gScan, 256, 0, stream>>>(S, a, f, out);
}
